// GraphSage_25271587570310
// MI455X (gfx1250) — matrix operand order repaired and checked by reading, not compiled, not run on hardware
//
#include <hip/hip_runtime.h>

#define NN 65536
#define EE 524288
#define CC 128
#define LL 3
#define APAD 136                      // 128 + 8 bf16 elems: kills LDS bank conflicts
#define SMEM_BYTES (3 * 128 * APAD * 2)

typedef __attribute__((ext_vector_type(16))) __bf16 v16bf;
typedef __attribute__((ext_vector_type(8)))  float  v8f;

union ABfrag { v16bf v; unsigned int u[8]; };

__device__ __forceinline__ unsigned short f2bf(float f) {
    unsigned int u = __float_as_uint(f);
    u += 0x7fffu + ((u >> 16) & 1u);          // round-to-nearest-even
    return (unsigned short)(u >> 16);
}

__device__ __forceinline__ unsigned int pack2bf(float lo, float hi) {
    return (unsigned int)f2bf(lo) | ((unsigned int)f2bf(hi) << 16);
}

// ---------------- edge prep: ew = log1p(w), deg = #in-edges -----------------
__global__ __launch_bounds__(256) void edge_prep(const float* __restrict__ w,
                                                 const int* __restrict__ dst,
                                                 float* __restrict__ ew,
                                                 float* __restrict__ deg) {
    int e = blockIdx.x * 256 + threadIdx.x;
    ew[e] = log1pf(w[e]);
    atomicAdd(&deg[dst[e]], 1.0f);
}

// ---- per-layer weight conversion: W[k][n] f32 -> packed bf16 [n][k-pairs] --
__global__ __launch_bounds__(256) void conv_weights(const float* __restrict__ Wself,
                                                    const float* __restrict__ Wneigh,
                                                    unsigned int* __restrict__ WbfS,
                                                    unsigned int* __restrict__ WbfN) {
    int i = blockIdx.x * 256 + threadIdx.x;   // over 128*64 u32 words
    int n = i >> 6, k2 = (i & 63) * 2;
    WbfS[i] = pack2bf(Wself [(size_t)k2 * CC + n], Wself [(size_t)(k2 + 1) * CC + n]);
    WbfN[i] = pack2bf(Wneigh[(size_t)k2 * CC + n], Wneigh[(size_t)(k2 + 1) * CC + n]);
}

// ---------------- BatchNorm column stats (sum, sumsq) -----------------------
__global__ __launch_bounds__(128) void bn_reduce(const float* __restrict__ h,
                                                 float* __restrict__ sums) {
    const int c  = threadIdx.x;               // channel
    const int r0 = blockIdx.x * 256;          // 256 rows per block
    float s = 0.f, q = 0.f;
    for (int r = 0; r < 256; ++r) {
        float v = h[(size_t)(r0 + r) * CC + c];
        s += v; q += v * v;
    }
    atomicAdd(&sums[c], s);
    atomicAdd(&sums[CC + c], q);
}

__global__ __launch_bounds__(128) void bn_finalize(const float* __restrict__ sums,
                                                   const float* __restrict__ gamma,
                                                   const float* __restrict__ beta,
                                                   float* __restrict__ scale,
                                                   float* __restrict__ shift) {
    int c = threadIdx.x;
    float mu  = sums[c] * (1.0f / (float)NN);
    float var = sums[CC + c] * (1.0f / (float)NN) - mu * mu;
    float sc  = gamma[c] * rsqrtf(var + 1e-5f);
    scale[c] = sc;
    shift[c] = beta[c] - mu * sc;
}

// BN apply: writes f32 hbn (for scatter) AND packed-bf16 hbn (GEMM A operand).
__global__ __launch_bounds__(256) void bn_apply(const float* __restrict__ h,
                                                const float* __restrict__ scale,
                                                const float* __restrict__ shift,
                                                float* __restrict__ o,
                                                unsigned int* __restrict__ obf) {
    size_t i = (size_t)blockIdx.x * 256 + threadIdx.x;     // over N*C/4 float4s
    int c4 = (int)(i & 31);                                 // channel group
    float4 v  = ((const float4*)h)[i];
    float4 sc = ((const float4*)scale)[c4];
    float4 sh = ((const float4*)shift)[c4];
    float4 r;
    r.x = v.x * sc.x + sh.x;  r.y = v.y * sc.y + sh.y;
    r.z = v.z * sc.z + sh.z;  r.w = v.w * sc.w + sh.w;
    ((float4*)o)[i] = r;
    uint2 p;
    p.x = pack2bf(r.x, r.y);
    p.y = pack2bf(r.z, r.w);
    ((uint2*)obf)[i] = p;
}

// ---------------- weighted scatter-add: neigh[dst] += hbn[src] * ew ---------
// One wave per edge; 32 lanes x float4 = 128 channels. L2-resident atomics.
__global__ __launch_bounds__(256) void scatter_edges(const float* __restrict__ hbn,
                                                     const float* __restrict__ ew,
                                                     const int* __restrict__ src,
                                                     const int* __restrict__ dst,
                                                     float* __restrict__ neigh) {
    const int wave = threadIdx.x >> 5;
    const int lane = threadIdx.x & 31;
    const int e = blockIdx.x * 8 + wave;                    // EE % 8 == 0
    const float w = ew[e];
    const int s = src[e], d = dst[e];
    float4 v = ((const float4*)(hbn + (size_t)s * CC))[lane];
    float* np = neigh + (size_t)d * CC + lane * 4;
    atomicAdd(np + 0, v.x * w);
    atomicAdd(np + 1, v.y * w);
    atomicAdd(np + 2, v.z * w);
    atomicAdd(np + 3, v.w * w);
}

// ---------------- fused dual-GEMM + bias + ReLU via WMMA bf16 ---------------
// out[128 rows] = relu(hbn @ Wself + (neigh/deg) @ Wneigh + bias)
// 8 waves: wave w owns rows w*16..w*16+15, all 8 col tiles (64 WMMAs/wave).
__global__ __launch_bounds__(256) void gemm_fused(const unsigned int* __restrict__ H1bf,
                                                  const float* __restrict__ neigh,
                                                  const float* __restrict__ deg,
                                                  const unsigned int* __restrict__ WbfS,
                                                  const unsigned int* __restrict__ WbfN,
                                                  const float* __restrict__ bias,
                                                  float* __restrict__ out) {
    extern __shared__ unsigned short smem[];
    unsigned short* Bself  = smem;                  // [128 cols][APAD] = W^T
    unsigned short* Bneigh = smem + 128 * APAD;
    unsigned short* As     = smem + 2 * 128 * APAD; // [128 rows][APAD]

    const int tid  = threadIdx.x;
    const int wave = tid >> 5;
    const int lane = tid & 31;
    const int half = lane >> 4;      // K-half selector (ISA layout)
    const int mn   = lane & 15;      // row (A) / col (B,D) within tile
    const int rowBase = blockIdx.x * 128;

    // Stage pre-converted weights + A tile: pure u32 copies, no conversion.
    unsigned int* Bs32 = (unsigned int*)Bself;
    unsigned int* Bn32 = (unsigned int*)Bneigh;
    unsigned int* As32 = (unsigned int*)As;
    for (int i = tid; i < 128 * 64; i += 256) {
        int n = i >> 6, k2 = i & 63;
        int l = n * (APAD >> 1) + k2;
        Bs32[l] = WbfS[i];
        Bn32[l] = WbfN[i];
        As32[l] = H1bf[((size_t)(rowBase + n) << 6) + k2];
    }
    __syncthreads();

    v8f acc[8];
#pragma unroll
    for (int t = 0; t < 8; ++t)
#pragma unroll
        for (int r = 0; r < 8; ++r) acc[t][r] = 0.0f;

    const unsigned int* ap = (const unsigned int*)As;
    const int arow = wave * 16 + mn;

    // ---- pass 1: acc += A_self x Wself ----
#pragma unroll
    for (int ks = 0; ks < 4; ++ks) {
        ABfrag a;
#pragma unroll
        for (int j = 0; j < 8; ++j) {
            int k0 = ks * 32 + ((j < 4) ? (2 * j + 8 * half)
                                        : (16 + 2 * (j - 4) + 8 * half));
            a.u[j] = ap[(arow * APAD + k0) >> 1];
        }
        const unsigned int* bp = (const unsigned int*)Bself;
#pragma unroll
        for (int tn = 0; tn < 8; ++tn) {
            ABfrag b;
            int col = tn * 16 + mn;
#pragma unroll
            for (int j = 0; j < 8; ++j) {
                int k = ks * 32 + half * 8 + 2 * j + ((j < 4) ? 0 : 8);
                b.u[j] = bp[(col * APAD + k) >> 1];
            }
            acc[tn] = __builtin_amdgcn_wmma_f32_16x16x32_bf16(
                false, a.v, false, b.v, (short)0, acc[tn], false, false);
        }
    }
    __syncthreads();

    // Restage A = neigh / max(deg,1): conversion unavoidable (f32 atomics).
    for (int i = tid; i < 128 * 64; i += 256) {
        int m = i >> 6, k2 = i & 63;
        int r = rowBase + m;
        float invd = 1.0f / fmaxf(deg[r], 1.0f);
        float2 v = ((const float2*)(neigh + (size_t)r * CC))[k2];
        As32[m * (APAD >> 1) + k2] = pack2bf(v.x * invd, v.y * invd);
    }
    __syncthreads();

    // ---- pass 2: acc += A_neigh x Wneigh ----
#pragma unroll
    for (int ks = 0; ks < 4; ++ks) {
        ABfrag a;
#pragma unroll
        for (int j = 0; j < 8; ++j) {
            int k0 = ks * 32 + ((j < 4) ? (2 * j + 8 * half)
                                        : (16 + 2 * (j - 4) + 8 * half));
            a.u[j] = ap[(arow * APAD + k0) >> 1];
        }
        const unsigned int* bp = (const unsigned int*)Bneigh;
#pragma unroll
        for (int tn = 0; tn < 8; ++tn) {
            ABfrag b;
            int col = tn * 16 + mn;
#pragma unroll
            for (int j = 0; j < 8; ++j) {
                int k = ks * 32 + half * 8 + 2 * j + ((j < 4) ? 0 : 8);
                b.u[j] = bp[(col * APAD + k) >> 1];
            }
            acc[tn] = __builtin_amdgcn_wmma_f32_16x16x32_bf16(
                false, a.v, false, b.v, (short)0, acc[tn], false, false);
        }
    }

    // Epilogue: bias + ReLU, D layout: VGPR r -> M = r + 8*half, N = mn.
#pragma unroll
    for (int tn = 0; tn < 8; ++tn) {
        int col = tn * 16 + mn;
        float bv = bias[col];
#pragma unroll
        for (int r = 0; r < 8; ++r) {
            int row = rowBase + wave * 16 + r + 8 * half;
            float v = acc[tn][r] + bv;
            out[(size_t)row * CC + col] = v > 0.0f ? v : 0.0f;
        }
    }
}

// ---------------- final linear: out[n] = h[n] . lin_w + lin_b ---------------
__global__ __launch_bounds__(256) void final_linear(const float* __restrict__ h,
                                                    const float* __restrict__ lw,
                                                    const float* __restrict__ lb,
                                                    float* __restrict__ out) {
    const int wave = threadIdx.x >> 5;
    const int lane = threadIdx.x & 31;
    const int row = blockIdx.x * 8 + wave;
    float4 v = ((const float4*)(h + (size_t)row * CC))[lane];
    float4 w = ((const float4*)lw)[lane];
    float p = v.x * w.x + v.y * w.y + v.z * w.z + v.w * w.w;
#pragma unroll
    for (int off = 16; off; off >>= 1) p += __shfl_down(p, off, 32);
    if (lane == 0) out[row] = p + lb[0];
}

extern "C" void kernel_launch(void* const* d_in, const int* in_sizes, int n_in,
                              void* d_out, int out_size, void* d_ws, size_t ws_size,
                              hipStream_t stream) {
    (void)in_sizes; (void)n_in; (void)out_size; (void)ws_size;
    const float* x       = (const float*)d_in[0];
    const float* edge_w  = (const float*)d_in[1];
    const float* bn_g    = (const float*)d_in[2];
    const float* bn_b    = (const float*)d_in[3];
    const float* W_self  = (const float*)d_in[4];
    const float* W_neigh = (const float*)d_in[5];
    const float* bias    = (const float*)d_in[6];
    const float* lin_w   = (const float*)d_in[7];
    const float* lin_b   = (const float*)d_in[8];
    const int*   src     = (const int*)d_in[9];
    const int*   dst     = (const int*)d_in[10];
    float* out = (float*)d_out;

    // Workspace (floats): ew[E] deg[N] scale[C] shift[C] sums[2C]
    //                     H1[N*C] NB[N*C] H2[N*C] | H1bf[N*C/2 u32] WbfS[8K] WbfN[8K]
    float* ws    = (float*)d_ws;
    float* ew    = ws;
    float* deg   = ew + EE;
    float* scale = deg + NN;
    float* shift = scale + CC;
    float* sums  = shift + CC;
    float* H1    = sums + 2 * CC;                 // hbn (normalized h, f32)
    float* NB    = H1 + (size_t)NN * CC;          // neighbor aggregate
    float* H2    = NB + (size_t)NN * CC;          // layer output
    unsigned int* H1bf = (unsigned int*)(H2 + (size_t)NN * CC);   // hbn bf16 packed
    unsigned int* WbfS = H1bf + (size_t)NN * CC / 2;
    unsigned int* WbfN = WbfS + CC * CC / 2;

    hipMemsetAsync(deg, 0, NN * sizeof(float), stream);
    edge_prep<<<EE / 256, 256, 0, stream>>>(edge_w, dst, ew, deg);

    const float* hin = x;
    for (int l = 0; l < LL; ++l) {
        conv_weights<<<(CC * CC / 2) / 256, 256, 0, stream>>>(
            W_self + (size_t)l * CC * CC, W_neigh + (size_t)l * CC * CC, WbfS, WbfN);
        hipMemsetAsync(sums, 0, 2 * CC * sizeof(float), stream);
        bn_reduce<<<NN / 256, 128, 0, stream>>>(hin, sums);
        bn_finalize<<<1, 128, 0, stream>>>(sums, bn_g + l * CC, bn_b + l * CC, scale, shift);
        bn_apply<<<(NN * CC / 4) / 256, 256, 0, stream>>>(hin, scale, shift, H1, H1bf);
        hipMemsetAsync(NB, 0, (size_t)NN * CC * sizeof(float), stream);
        scatter_edges<<<EE / 8, 256, 0, stream>>>(H1, ew, src, dst, NB);
        gemm_fused<<<NN / 128, 256, SMEM_BYTES, stream>>>(
            H1bf, NB, deg, WbfS, WbfN, bias + l * CC, H2);
        hin = H2;
    }
    final_linear<<<NN / 8, 256, 0, stream>>>(H2, lin_w, lin_b, out);
}
